// MoETransformer_47914655154647
// MI455X (gfx1250) — matrix operand order repaired and checked by reading, not compiled, not run on hardware
//
#include <hip/hip_runtime.h>
#include <hip/hip_bf16.h>

// Model dims (match reference)
#define DM 512
#define HH 8
#define DH 64
#define FF 2048
#define EE 8
#define LL 2
#define BB 4
#define SS 512
#define TT 512
#define VV 32000
#define NTOK (BB * SS)
#define NEGINF (-1e9f)

typedef __attribute__((ext_vector_type(16))) __bf16   v16bf;
typedef __attribute__((ext_vector_type(8)))  float    v8f;
typedef __attribute__((ext_vector_type(4)))  unsigned u4v;
typedef __attribute__((ext_vector_type(8)))  unsigned u8v;

// float -> bf16 (round-half-up), single element
__device__ __forceinline__ __bf16 f2bf(float f) {
  unsigned u = __builtin_bit_cast(unsigned, f) + 0x8000u;
  return __builtin_bit_cast(__bf16, (unsigned short)(u >> 16));
}

// two floats -> packed bf16 pair in one dword via v_perm_b32
__device__ __forceinline__ unsigned pack2bf(float f0, float f1) {
  unsigned u0 = __builtin_bit_cast(unsigned, f0) + 0x8000u;
  unsigned u1 = __builtin_bit_cast(unsigned, f1) + 0x8000u;
  // dst = { u1.b3, u1.b2, u0.b3, u0.b2 }  (S0=u1 -> codes 4..7, S1=u0 -> 0..3)
  return __builtin_amdgcn_perm(u1, u0, 0x07060302u);
}

// Build a WMMA bf16 fragment from two contiguous 16-byte LDS chunks
__device__ __forceinline__ v16bf frag_from_lds(const __bf16* p0, const __bf16* p1) {
  u4v lo = *(const u4v*)p0;
  u4v hi = *(const u4v*)p1;
  u8v c = {lo.x, lo.y, lo.z, lo.w, hi.x, hi.y, hi.z, hi.w};
  return __builtin_bit_cast(v16bf, c);
}

// ---------------------------------------------------------------------------
// LDS-tiled NN GEMM: C[M,N] = post(A[M,K] @ W[K,N] + bias)
//   block = 256 threads (8 waves), macro-tile 128(M) x 64(N), K-step 32
//   wave (waveM 0..3, waveN 0..1) computes 32x32 via 4 WMMAs / K-step
//   A staged [m][k] bf16, B staged TRANSPOSED [n][k] bf16 -> all fragment
//   reads are contiguous ds_load_b128 pairs.
// Requires: M % 128 == 0, N % 64 == 0, K % 32 == 0 (true for all call sites).
// ---------------------------------------------------------------------------
template <bool RELU, bool ACC, bool SCALE>
__global__ void __launch_bounds__(256)
gemm_nn_tiled(const float* __restrict__ A, const float* __restrict__ W,
              const float* __restrict__ bias, const float* __restrict__ rowScale,
              int rsStride, int rsOff, float* __restrict__ C,
              int M, int N, int K)
{
  __shared__ __attribute__((aligned(32))) __bf16 As[128 * 32];  // [m][k] 8KB
  __shared__ __attribute__((aligned(32))) __bf16 Bs[64 * 32];   // [n][k] 4KB

  const int tid   = threadIdx.x;
  const int lane  = tid & 31;
  const int wave  = tid >> 5;
  const int waveM = wave >> 1;                 // 0..3
  const int waveN = wave & 1;                  // 0..1
  const int mloc  = lane & 15;
  const int ka    = (lane < 16) ? 0 : 16;      // A frag K-base
  const int kb    = (lane < 16) ? 0 : 16;      // B frag K-base

  // staging coordinates
  const int arow  = tid >> 1;                  // 0..127
  const int ahalf = (tid & 1) * 16;            // 0 / 16
  const int bkk   = tid >> 3;                  // 0..31
  const int bc0   = (tid & 7) * 8;             // 0..56

  const float* aSrc = A + ((long)(blockIdx.y * 128 + arow)) * K + ahalf;
  const float* bSrc = W + (long)bkk * N + blockIdx.x * 64 + bc0;

  const v8f z = {0.f, 0.f, 0.f, 0.f, 0.f, 0.f, 0.f, 0.f};
  v8f acc[2][2] = {{z, z}, {z, z}};

  for (int k0 = 0; k0 < K; k0 += 32) {
    __syncthreads();   // previous iteration's fragment reads done

    // ---- stage A: 128x32 fp32 -> bf16, perm-packed, b128 LDS stores ----
    {
      const float* gp = aSrc + k0;
      __builtin_prefetch(gp + 32, 0, 1);
      float4 f0 = *(const float4*)(gp);
      float4 f1 = *(const float4*)(gp + 4);
      float4 f2 = *(const float4*)(gp + 8);
      float4 f3 = *(const float4*)(gp + 12);
      u4v w0 = {pack2bf(f0.x, f0.y), pack2bf(f0.z, f0.w),
                pack2bf(f1.x, f1.y), pack2bf(f1.z, f1.w)};
      u4v w1 = {pack2bf(f2.x, f2.y), pack2bf(f2.z, f2.w),
                pack2bf(f3.x, f3.y), pack2bf(f3.z, f3.w)};
      unsigned* dst = (unsigned*)(As + arow * 32 + ahalf);
      *(u4v*)dst       = w0;
      *(u4v*)(dst + 4) = w1;
    }
    // ---- stage B: 32x64 fp32, transposed into [n][k] bf16 ----
    {
      const float* gp = bSrc + (long)k0 * N;
      __builtin_prefetch(gp + (long)32 * N, 0, 1);
      float4 f0 = *(const float4*)(gp);
      float4 f1 = *(const float4*)(gp + 4);
      Bs[(bc0 + 0) * 32 + bkk] = f2bf(f0.x);
      Bs[(bc0 + 1) * 32 + bkk] = f2bf(f0.y);
      Bs[(bc0 + 2) * 32 + bkk] = f2bf(f0.z);
      Bs[(bc0 + 3) * 32 + bkk] = f2bf(f0.w);
      Bs[(bc0 + 4) * 32 + bkk] = f2bf(f1.x);
      Bs[(bc0 + 5) * 32 + bkk] = f2bf(f1.y);
      Bs[(bc0 + 6) * 32 + bkk] = f2bf(f1.z);
      Bs[(bc0 + 7) * 32 + bkk] = f2bf(f1.w);
    }
    __syncthreads();   // tiles visible

    // ---- compute: 2 A frags x 2 B frags -> 4 WMMAs ----
    v16bf af[2], bfr[2];
#pragma unroll
    for (int sm = 0; sm < 2; ++sm) {
      const __bf16* p = As + (waveM * 32 + sm * 16 + mloc) * 32 + ka;
      af[sm] = frag_from_lds(p, p + 8);
    }
#pragma unroll
    for (int sn = 0; sn < 2; ++sn) {
      const __bf16* p = Bs + (waveN * 32 + sn * 16 + mloc) * 32 + kb;
      bfr[sn] = frag_from_lds(p, p + 8);
    }
#pragma unroll
    for (int sm = 0; sm < 2; ++sm)
#pragma unroll
      for (int sn = 0; sn < 2; ++sn)
        acc[sm][sn] = __builtin_amdgcn_wmma_f32_16x16x32_bf16(
            false, af[sm], false, bfr[sn], (short)0, acc[sm][sn], false, false);
  }

  // ---- branch-free epilogue (template-specialized) ----
  const int baseM = blockIdx.y * 128 + waveM * 32;
  const int baseN = blockIdx.x * 64 + waveN * 32;
  const int rsub  = (lane < 16) ? 0 : 8;
#pragma unroll
  for (int sm = 0; sm < 2; ++sm) {
#pragma unroll
    for (int sn = 0; sn < 2; ++sn) {
      const int   col = baseN + sn * 16 + mloc;
      const float bsv = bias[col];
      const int   mb  = baseM + sm * 16 + rsub;
#pragma unroll
      for (int r = 0; r < 8; ++r) {
        const int mm = mb + r;
        float v = acc[sm][sn][r] + bsv;
        if (RELU) v = fmaxf(v, 0.f);
        if (SCALE) v *= rowScale[(long)mm * rsStride + rsOff];
        const long off = (long)mm * N + col;
        if (ACC) C[off] += v; else C[off] = v;
      }
    }
  }
}

// ---------------------------------------------------------------------------
// Attention logits: logits[z=(b,h)][q,k] = (Q_bh[q,:] . K_bh[k,:]) / 8
//   + pad mask (keyIds[b,k]==0) + optional causal (k > q).  NT WMMA, K=64.
// ---------------------------------------------------------------------------
__global__ void __launch_bounds__(128)
attn_qk_wmma(const float* __restrict__ Q, const float* __restrict__ Kb,
             const int* __restrict__ keyIds, float* __restrict__ logits,
             int Tq, int Skv, int causal)
{
  const int z = blockIdx.z, b = z >> 3, h = z & 7;
  const int lane = threadIdx.x & 31, wave = threadIdx.x >> 5;
  const int tileN = blockIdx.x * 4 + wave;
  const int tileM = blockIdx.y;

  const float* Aq = Q  + (long)b * Tq  * DM + h * DH;
  const float* Bk = Kb + (long)b * Skv * DM + h * DH;
  float*       Cp = logits + (long)z * Tq * Skv;

  const int  mloc = lane & 15;
  const int  ka   = (lane < 16) ? 0 : 16;
  const int  kb2  = (lane < 16) ? 0 : 16;
  const long rowA = (long)tileM * 16 + mloc;
  const long rowB = (long)tileN * 16 + mloc;

  v8f acc = {0.f, 0.f, 0.f, 0.f, 0.f, 0.f, 0.f, 0.f};
#pragma unroll
  for (int k0 = 0; k0 < DH; k0 += 32) {
    const float* ap = Aq + rowA * DM + k0 + ka;
    const float* bp = Bk + rowB * DM + k0 + kb2;
    v16bf afrag, bfrag;
#pragma unroll
    for (int e = 0; e < 8; ++e) { afrag[e] = f2bf(ap[e]); afrag[e + 8] = f2bf(ap[e + 8]); }
#pragma unroll
    for (int e = 0; e < 16; ++e) bfrag[e] = f2bf(bp[e]);   // NT: contiguous in K
    acc = __builtin_amdgcn_wmma_f32_16x16x32_bf16(false, afrag, false, bfrag,
                                                  (short)0, acc, false, false);
  }

  const int  n   = tileN * 16 + mloc;
  const int  mb  = tileM * 16 + ((lane < 16) ? 0 : 8);
  const bool pad = (keyIds[(long)b * Skv + n] == 0);
#pragma unroll
  for (int r = 0; r < 8; ++r) {
    const int mm = mb + r;
    float v = acc[r] * 0.125f;                 // 1/sqrt(DH)
    if (pad || (causal && n > mm)) v += NEGINF;
    Cp[(long)mm * Skv + n] = v;
  }
}

// ---------------------------------------------------------------------------
// O_bh[q,d] = sum_k att[z][q,k] * V_bh[k,d]; writes head-interleaved [token,DM]
// ---------------------------------------------------------------------------
__global__ void __launch_bounds__(128)
attn_av_wmma(const float* __restrict__ Att, const float* __restrict__ Vb,
             float* __restrict__ O, int Tq, int Skv)
{
  const int z = blockIdx.z, b = z >> 3, h = z & 7;
  const int lane = threadIdx.x & 31, wave = threadIdx.x >> 5;
  const int tileN = wave;                      // N = DH = 64 -> 4 tiles
  const int tileM = blockIdx.y;

  const float* Ap = Att + (long)z * Tq * Skv;
  const float* Bp = Vb + (long)b * Skv * DM + h * DH;

  const int  mloc = lane & 15;
  const int  ka   = (lane < 16) ? 0 : 16;
  const int  kb2  = (lane < 16) ? 0 : 16;
  const long rowA = (long)tileM * 16 + mloc;
  const int  colB = tileN * 16 + mloc;

  v8f acc = {0.f, 0.f, 0.f, 0.f, 0.f, 0.f, 0.f, 0.f};
  for (int k0 = 0; k0 < Skv; k0 += 32) {
    const float* ap = Ap + rowA * (long)Skv + k0 + ka;
    const float* bp = Bp + (long)(k0 + kb2) * DM + colB;
    v16bf afrag, bfrag;
#pragma unroll
    for (int e = 0; e < 8; ++e) { afrag[e] = f2bf(ap[e]); afrag[e + 8] = f2bf(ap[e + 8]); }
#pragma unroll
    for (int e = 0; e < 16; ++e) bfrag[e] = f2bf(bp[(long)e * DM]);
    acc = __builtin_amdgcn_wmma_f32_16x16x32_bf16(false, afrag, false, bfrag,
                                                  (short)0, acc, false, false);
  }

  const int mb = tileM * 16 + ((lane < 16) ? 0 : 8);
#pragma unroll
  for (int r = 0; r < 8; ++r) {
    const int mm = mb + r;
    O[((long)b * Tq + mm) * DM + h * DH + colB] = acc[r];
  }
}

// ---------------------------------------------------------------------------
// Row softmax over length-N rows (in place). One block per row.
// ---------------------------------------------------------------------------
__global__ void __launch_bounds__(256)
softmax_rows(float* __restrict__ x, int N)
{
  const long row = blockIdx.x;
  float* p = x + row * (long)N;
  __shared__ float red[256];
  const int tid = threadIdx.x;

  float mx = -3.0e38f;
  for (int i = tid; i < N; i += 256) mx = fmaxf(mx, p[i]);
  red[tid] = mx; __syncthreads();
  for (int s = 128; s > 0; s >>= 1) { if (tid < s) red[tid] = fmaxf(red[tid], red[tid + s]); __syncthreads(); }
  mx = red[0]; __syncthreads();

  float sum = 0.f;
  for (int i = tid; i < N; i += 256) { float e = __expf(p[i] - mx); p[i] = e; sum += e; }
  red[tid] = sum; __syncthreads();
  for (int s = 128; s > 0; s >>= 1) { if (tid < s) red[tid] += red[tid + s]; __syncthreads(); }
  const float inv = 1.f / red[0];
  for (int i = tid; i < N; i += 256) p[i] *= inv;
}

// ---------------------------------------------------------------------------
// out = LN(x + a) * g + b   (D = 512 fixed; one block of 256 per row; in-place OK)
// ---------------------------------------------------------------------------
__global__ void __launch_bounds__(256)
add_ln_kernel(const float* __restrict__ xin, const float* __restrict__ add,
              const float* __restrict__ g, const float* __restrict__ b,
              float* __restrict__ out)
{
  const long row = blockIdx.x;
  const int  tid = threadIdx.x;
  const float v0 = xin[row * DM + tid]       + add[row * DM + tid];
  const float v1 = xin[row * DM + tid + 256] + add[row * DM + tid + 256];
  __shared__ float red[256];

  red[tid] = v0 + v1; __syncthreads();
  for (int s = 128; s > 0; s >>= 1) { if (tid < s) red[tid] += red[tid + s]; __syncthreads(); }
  const float mean = red[0] * (1.f / DM);
  __syncthreads();

  const float d0 = v0 - mean, d1 = v1 - mean;
  red[tid] = d0 * d0 + d1 * d1; __syncthreads();
  for (int s = 128; s > 0; s >>= 1) { if (tid < s) red[tid] += red[tid + s]; __syncthreads(); }
  const float inv = rsqrtf(red[0] * (1.f / DM) + 1e-6f);

  out[row * DM + tid]       = d0 * inv * g[tid]       + b[tid];
  out[row * DM + tid + 256] = d1 * inv * g[tid + 256] + b[tid + 256];
}

// ---------------------------------------------------------------------------
// Embedding gather + sinusoidal positional encoding (computed on the fly)
// ---------------------------------------------------------------------------
__global__ void __launch_bounds__(256)
embed_kernel(const int* __restrict__ ids, const float* __restrict__ emb,
             float* __restrict__ out, int seqLen)
{
  const long t   = blockIdx.x;
  const int  pos = (int)(t % seqLen);
  const int  id  = ids[t];
  const int  tid = threadIdx.x;
  const float k  = 9.210340371976184f / 256.f;   // ln(10000)/half
  for (int d = tid; d < DM; d += 256) {
    float pe;
    if (d < 256) { float r = __expf(-(float)d * k);          pe = __sinf((float)pos * r); }
    else         { float r = __expf(-(float)(d - 256) * k);  pe = __cosf((float)pos * r); }
    out[t * DM + d] = emb[(long)id * DM + d] + pe;
  }
}

// ---------------------------------------------------------------------------
// MoE gate: gates[t,:] = softmax(x[t,:] @ gw + gb), E = 8
// ---------------------------------------------------------------------------
__global__ void __launch_bounds__(256)
moe_gate_kernel(const float* __restrict__ x, const float* __restrict__ gw,
                const float* __restrict__ gb, float* __restrict__ gates)
{
  __shared__ float red[256 * EE];
  const long t   = blockIdx.x;
  const int  tid = threadIdx.x;
  float loc[EE];
#pragma unroll
  for (int e = 0; e < EE; ++e) loc[e] = 0.f;
  for (int d = tid; d < DM; d += 256) {
    const float xv = x[t * DM + d];
#pragma unroll
    for (int e = 0; e < EE; ++e) loc[e] += xv * gw[d * EE + e];
  }
#pragma unroll
  for (int e = 0; e < EE; ++e) red[tid * EE + e] = loc[e];
  __syncthreads();
  for (int s = 128; s > 0; s >>= 1) {
    if (tid < s) {
#pragma unroll
      for (int e = 0; e < EE; ++e) red[tid * EE + e] += red[(tid + s) * EE + e];
    }
    __syncthreads();
  }
  if (tid == 0) {
    float lg[EE], mx = -3.0e38f, sum = 0.f;
#pragma unroll
    for (int e = 0; e < EE; ++e) { lg[e] = red[e] + gb[e]; mx = fmaxf(mx, lg[e]); }
#pragma unroll
    for (int e = 0; e < EE; ++e) { lg[e] = __expf(lg[e] - mx); sum += lg[e]; }
    const float inv = 1.f / sum;
#pragma unroll
    for (int e = 0; e < EE; ++e) gates[t * EE + e] = lg[e] * inv;
  }
}

// ===========================================================================
extern "C" void kernel_launch(void* const* d_in, const int* in_sizes, int n_in,
                              void* d_out, int out_size, void* d_ws, size_t ws_size,
                              hipStream_t stream)
{
  (void)in_sizes; (void)n_in; (void)out_size; (void)ws_size;

  const int*   inp     = (const int*)d_in[0];
  const int*   tar     = (const int*)d_in[1];
  const float* emb_enc = (const float*)d_in[2];
  const float* emb_dec = (const float*)d_in[3];
  // enc_mha
  const float *e_wq = (const float*)d_in[4],  *e_bq = (const float*)d_in[5];
  const float *e_wk = (const float*)d_in[6],  *e_bk = (const float*)d_in[7];
  const float *e_wv = (const float*)d_in[8],  *e_bv = (const float*)d_in[9];
  const float *e_wo = (const float*)d_in[10], *e_bo = (const float*)d_in[11];
  // enc_ffn
  const float *e_f1  = (const float*)d_in[12], *e_fb1 = (const float*)d_in[13];
  const float *e_f2  = (const float*)d_in[14], *e_fb2 = (const float*)d_in[15];
  // enc_ln
  const float *e_g1 = (const float*)d_in[16], *e_b1 = (const float*)d_in[17];
  const float *e_g2 = (const float*)d_in[18], *e_b2 = (const float*)d_in[19];
  // moe
  const float *m_gw  = (const float*)d_in[20], *m_gb  = (const float*)d_in[21];
  const float *m_w1  = (const float*)d_in[22], *m_b1  = (const float*)d_in[23];
  const float *m_w2  = (const float*)d_in[24], *m_b2  = (const float*)d_in[25];
  const float *m_lng = (const float*)d_in[26], *m_lnb = (const float*)d_in[27];
  // dec_mha1
  const float *q1_wq = (const float*)d_in[28], *q1_bq = (const float*)d_in[29];
  const float *q1_wk = (const float*)d_in[30], *q1_bk = (const float*)d_in[31];
  const float *q1_wv = (const float*)d_in[32], *q1_bv = (const float*)d_in[33];
  const float *q1_wo = (const float*)d_in[34], *q1_bo = (const float*)d_in[35];
  // dec_mha2
  const float *q2_wq = (const float*)d_in[36], *q2_bq = (const float*)d_in[37];
  const float *q2_wk = (const float*)d_in[38], *q2_bk = (const float*)d_in[39];
  const float *q2_wv = (const float*)d_in[40], *q2_bv = (const float*)d_in[41];
  const float *q2_wo = (const float*)d_in[42], *q2_bo = (const float*)d_in[43];
  // dec_ffn
  const float *d_f1  = (const float*)d_in[44], *d_fb1 = (const float*)d_in[45];
  const float *d_f2  = (const float*)d_in[46], *d_fb2 = (const float*)d_in[47];
  // dec_ln
  const float *g1 = (const float*)d_in[48], *b1 = (const float*)d_in[49];
  const float *g2 = (const float*)d_in[50], *b2 = (const float*)d_in[51];
  const float *g3 = (const float*)d_in[52], *b3 = (const float*)d_in[53];
  const float *fin_w = (const float*)d_in[54], *fin_b = (const float*)d_in[55];
  float* out = (float*)d_out;

  // --- workspace carve (~84 MB of floats) ---
  float* ws = (float*)d_ws;
  size_t off = 0;
  auto carve = [&](size_t n) { float* p = ws + off; off += n; return p; };
  float* x      = carve((size_t)NTOK * DM);   // encoder stream / enc_out
  float* y      = carve((size_t)NTOK * DM);   // decoder stream
  float* qb     = carve((size_t)NTOK * DM);
  float* kb     = carve((size_t)NTOK * DM);
  float* vb     = carve((size_t)NTOK * DM);
  float* attno  = carve((size_t)NTOK * DM);
  float* proj   = carve((size_t)NTOK * DM);
  float* logits = carve((size_t)BB * HH * TT * SS);
  float* hbuf   = carve((size_t)NTOK * FF);
  float* moeacc = carve((size_t)NTOK * DM);
  float* gates  = carve((size_t)NTOK * EE);

  // template-dispatched tiled WMMA GEMM (M%128==0, N%64==0, K%32==0)
  auto gemm = [&](const float* A, const float* W, const float* bias, float* C,
                  int M, int N, int K, bool relu, bool accum,
                  const float* rs, int rsOff) {
    dim3 g(N / 64, M / 128, 1);
    if (rs) {
      if (accum)
        gemm_nn_tiled<false, true,  true ><<<g, dim3(256), 0, stream>>>(A, W, bias, rs, EE, rsOff, C, M, N, K);
      else
        gemm_nn_tiled<false, false, true ><<<g, dim3(256), 0, stream>>>(A, W, bias, rs, EE, rsOff, C, M, N, K);
    } else if (relu) {
      gemm_nn_tiled<true,  false, false><<<g, dim3(256), 0, stream>>>(A, W, bias, nullptr, 0, 0, C, M, N, K);
    } else {
      gemm_nn_tiled<false, false, false><<<g, dim3(256), 0, stream>>>(A, W, bias, nullptr, 0, 0, C, M, N, K);
    }
  };

  auto mha = [&](const float* xq, const float* xkv, int rowsQ, int rowsKV,
                 const float* wq, const float* bq, const float* wk, const float* bk,
                 const float* wv, const float* bv, const float* wo, const float* bo,
                 const int* keyIds, int causal, float* outp) {
    gemm(xq,  wq, bq, qb, rowsQ,  DM, DM, false, false, nullptr, 0);
    gemm(xkv, wk, bk, kb, rowsKV, DM, DM, false, false, nullptr, 0);
    gemm(xkv, wv, bv, vb, rowsKV, DM, DM, false, false, nullptr, 0);
    const int Tq = rowsQ / BB, Skv = rowsKV / BB;
    attn_qk_wmma<<<dim3(Skv / 64, Tq / 16, BB * HH), dim3(128), 0, stream>>>(
        qb, kb, keyIds, logits, Tq, Skv, causal);
    softmax_rows<<<dim3(BB * HH * Tq), dim3(256), 0, stream>>>(logits, Skv);
    attn_av_wmma<<<dim3(1, Tq / 16, BB * HH), dim3(128), 0, stream>>>(
        logits, vb, attno, Tq, Skv);
    gemm(attno, wo, bo, outp, rowsQ, DM, DM, false, false, nullptr, 0);
  };

  auto ln = [&](const float* xin, const float* addv, const float* gg,
                const float* bb, float* o2, int rows) {
    add_ln_kernel<<<dim3(rows), dim3(256), 0, stream>>>(xin, addv, gg, bb, o2);
  };

  // ===================== encoder =====================
  embed_kernel<<<dim3(NTOK), dim3(256), 0, stream>>>(inp, emb_enc, x, SS);
  for (int l = 0; l < LL; ++l) {
    mha(x, x, NTOK, NTOK,
        e_wq + (size_t)l * DM * DM, e_bq + (size_t)l * DM,
        e_wk + (size_t)l * DM * DM, e_bk + (size_t)l * DM,
        e_wv + (size_t)l * DM * DM, e_bv + (size_t)l * DM,
        e_wo + (size_t)l * DM * DM, e_bo + (size_t)l * DM,
        inp, 0, proj);
    ln(x, proj, e_g1 + (size_t)l * DM, e_b1 + (size_t)l * DM, x, NTOK);

    gemm(x, e_f1 + (size_t)l * DM * FF, e_fb1 + (size_t)l * FF, hbuf, NTOK, FF, DM, true, false, nullptr, 0);
    gemm(hbuf, e_f2 + (size_t)l * FF * DM, e_fb2 + (size_t)l * DM, proj, NTOK, DM, FF, false, false, nullptr, 0);
    ln(x, proj, e_g2 + (size_t)l * DM, e_b2 + (size_t)l * DM, x, NTOK);

    // MoE: dense mixture over E experts, gate-weighted accumulation in epilogue
    moe_gate_kernel<<<dim3(NTOK), dim3(256), 0, stream>>>(
        x, m_gw + (size_t)l * DM * EE, m_gb + (size_t)l * EE, gates);
    for (int e = 0; e < EE; ++e) {
      const float* w1p = m_w1 + ((size_t)(l * EE + e)) * DM * FF;
      const float* b1p = m_b1 + ((size_t)(l * EE + e)) * FF;
      const float* w2p = m_w2 + ((size_t)(l * EE + e)) * FF * DM;
      const float* b2p = m_b2 + ((size_t)(l * EE + e)) * DM;
      gemm(x, w1p, b1p, hbuf, NTOK, FF, DM, true, false, nullptr, 0);
      gemm(hbuf, w2p, b2p, moeacc, NTOK, DM, FF, false, (e != 0), gates, e);
    }
    ln(x, moeacc, m_lng + (size_t)l * DM, m_lnb + (size_t)l * DM, x, NTOK);
  }
  // x now holds enc_out

  // ===================== decoder =====================
  embed_kernel<<<dim3(NTOK), dim3(256), 0, stream>>>(tar, emb_dec, y, TT);
  for (int l = 0; l < LL; ++l) {
    // masked self-attention (pad + causal)
    mha(y, y, NTOK, NTOK,
        q1_wq + (size_t)l * DM * DM, q1_bq + (size_t)l * DM,
        q1_wk + (size_t)l * DM * DM, q1_bk + (size_t)l * DM,
        q1_wv + (size_t)l * DM * DM, q1_bv + (size_t)l * DM,
        q1_wo + (size_t)l * DM * DM, q1_bo + (size_t)l * DM,
        tar, 1, proj);
    ln(y, proj, g1 + (size_t)l * DM, b1 + (size_t)l * DM, y, NTOK);

    // cross-attention against enc_out (pad mask from inp)
    mha(y, x, NTOK, NTOK,
        q2_wq + (size_t)l * DM * DM, q2_bq + (size_t)l * DM,
        q2_wk + (size_t)l * DM * DM, q2_bk + (size_t)l * DM,
        q2_wv + (size_t)l * DM * DM, q2_bv + (size_t)l * DM,
        q2_wo + (size_t)l * DM * DM, q2_bo + (size_t)l * DM,
        inp, 0, proj);
    ln(y, proj, g2 + (size_t)l * DM, b2 + (size_t)l * DM, y, NTOK);

    gemm(y, d_f1 + (size_t)l * DM * FF, d_fb1 + (size_t)l * FF, hbuf, NTOK, FF, DM, true, false, nullptr, 0);
    gemm(hbuf, d_f2 + (size_t)l * FF * DM, d_fb2 + (size_t)l * DM, proj, NTOK, DM, FF, false, false, nullptr, 0);
    ln(y, proj, g3 + (size_t)l * DM, b3 + (size_t)l * DM, y, NTOK);
  }

  // final vocab projection: [NTOK, DM] @ [DM, VV] + bias -> d_out
  gemm(y, fin_w, fin_b, out, NTOK, VV, DM, false, false, nullptr, 0);
}
